// AgentNetworkWHypernet_67233418051918
// MI455X (gfx1250) — matrix operand order repaired and checked by reading, not compiled, not run on hardware
//
#include <hip/hip_runtime.h>

typedef __bf16 bf16_t;
typedef __attribute__((ext_vector_type(16))) __bf16 v16bf;
typedef __attribute__((ext_vector_type(4)))  __bf16 v4bf;
typedef __attribute__((ext_vector_type(8)))  float  v8f;
typedef __attribute__((ext_vector_type(4)))  unsigned int u32x4;
typedef __attribute__((ext_vector_type(4)))  int i32x4;
typedef __attribute__((ext_vector_type(8)))  int i32x8;

#define BN    2048   // 512*4 samples
#define E     128    // hypernet embed
#define HID   1280   // 10*E
#define INF   192    // input feature length (OBS+OBSU)
#define IDF   16
#define STATE 176

// ---------------------------------------------------------------------------
// h = relu(ids @ l1_W.T + l1_b)   [BN, 1280]
// ---------------------------------------------------------------------------
__global__ void prep_h(const float* __restrict__ input, const float* __restrict__ l1W,
                       const float* __restrict__ l1b, float* __restrict__ h) {
  int idx = blockIdx.x * blockDim.x + threadIdx.x;
  if (idx >= BN * HID) return;
  int b = idx / HID, j = idx % HID;
  const float* ids = input + (size_t)b * INF + STATE;
  const float* wr  = l1W + (size_t)j * IDF;
  float acc = l1b[j];
#pragma unroll
  for (int k = 0; k < IDF; ++k) acc += ids[k] * wr[k];
  h[idx] = acc > 0.f ? acc : 0.f;
}

// extract x0 = input[..., :STATE]  [BN, STATE]
__global__ void prep_x0(const float* __restrict__ input, float* __restrict__ x0) {
  int idx = blockIdx.x * blockDim.x + threadIdx.x;
  if (idx >= BN * STATE) return;
  int b = idx / STATE, s = idx % STATE;
  x0[idx] = input[(size_t)b * INF + s];
}

// fp32 -> bf16 bulk convert (4 elements per thread)
__global__ void cvt_bf16(const float* __restrict__ src, bf16_t* __restrict__ dst, int n4) {
  int i = blockIdx.x * blockDim.x + threadIdx.x;
  if (i >= n4) return;
  const float4 f = ((const float4*)src)[i];
  v4bf o;
  o[0] = (bf16_t)f.x; o[1] = (bf16_t)f.y; o[2] = (bf16_t)f.z; o[3] = (bf16_t)f.w;
  ((v4bf*)dst)[i] = o;
}

// ---------------------------------------------------------------------------
// BM[b,m] = bb[m] + h_ib[b,:] . bW[m,:] + sum_s x[b,s] * wb[s*OUT+m]
// ---------------------------------------------------------------------------
template <int IN, int OUT>
__global__ void bias_mat(const float* __restrict__ x, const float* __restrict__ hck,
                         const float* __restrict__ wb, const float* __restrict__ bW,
                         const float* __restrict__ bb, float* __restrict__ BM) {
  int idx = blockIdx.x * blockDim.x + threadIdx.x;
  if (idx >= BN * OUT) return;
  int b = idx / OUT, m = idx % OUT;
  float acc = bb[m];
  const float* hr  = hck + (size_t)b * HID;
  const float* bWr = bW + (size_t)m * E;
  for (int e = 0; e < E; ++e) acc += hr[e] * bWr[e];
  const float* xr = x + (size_t)b * IN;
  for (int s = 0; s < IN; ++s) acc += xr[s] * wb[(size_t)s * OUT + m];
  BM[idx] = acc;
}

// ---------------------------------------------------------------------------
// Main fused hyper-layer: y[b,m] = relu?( BM[b,m] + sum_{s,e} x[b,s]h[b,e]wW[s*OUT+m,e] )
// B matrix streamed by the Tensor Data Mover into double-buffered LDS
// (row-padded for bank-conflict-free ds_load_b128), one K-iteration ahead.
// One wave = 16(sample) x 16*NT(out) tile; bf16 WMMA, K = IN*128.
// ---------------------------------------------------------------------------
template <int IN, int OUT, int RELU, int NW, int NT>
__global__ __launch_bounds__(NW * 32)
void layer_wmma(const float* __restrict__ x, const float* __restrict__ hck,
                const bf16_t* __restrict__ wbf, const float* __restrict__ BM,
                float* __restrict__ y) {
  constexpr int COLS  = NW * NT * 16;   // output columns per block
  constexpr int ROWB  = 272;            // 256B row + 16B pad (4-bank col stride)
  constexpr int TILEB = COLS * ROWB;    // one B stage tile

  __shared__ __align__(16) char  bstage[2][TILEB];
  __shared__ float xs[IN * 16];         // x tile, transposed [s][row]

  const int b0    = blockIdx.x * 16;
  const int wave  = (int)(threadIdx.x >> 5);
  const int lane  = (int)(threadIdx.x & 31);
  const int n0b   = blockIdx.y * COLS;
  const int n0    = n0b + wave * (16 * NT);
  const int row   = lane & 15;          // sample row for A / out column for B,C
  const int khalf = lane >> 4;          // which K half this lane holds

  for (int i = threadIdx.x; i < IN * 16; i += NW * 32) {
    int s = i >> 4, r = i & 15;
    xs[i] = x[(size_t)(b0 + r) * IN + s];
  }

  // h fragment in A-operand layout: lanes<16 hold K 0-15, lanes>=16 hold K 16-31
  v16bf hf[4];
  {
    const float* hp = hck + (size_t)(b0 + row) * HID + khalf * 16;
#pragma unroll
    for (int c = 0; c < 4; ++c) {
#pragma unroll
      for (int j = 0; j < 8; ++j) {
        hf[c][j]     = (bf16_t)hp[c * 32 + j];
        hf[c][j + 8] = (bf16_t)hp[c * 32 + 8 + j];
      }
    }
  }

  // accumulators init from bias matrix (C layout: vgpr v -> M = v + 8*khalf, N = lane&15)
  v8f acc[NT];
#pragma unroll
  for (int t = 0; t < NT; ++t)
#pragma unroll
    for (int v = 0; v < 8; ++v)
      acc[t][v] = BM[(size_t)(b0 + v + khalf * 8) * OUT + (n0 + t * 16 + row)];

  // --- Tensor DMA descriptor (D#), ISA ch.8: 2D tile 128(e) x COLS(rows) ---
  const unsigned lds0 = (unsigned)(uintptr_t)&bstage[0][0];
  const unsigned lds1 = (unsigned)(uintptr_t)&bstage[1][0];
  unsigned long long gaddr =
      (unsigned long long)(uintptr_t)(wbf + (size_t)n0b * E);       // tile @ s=0
  const unsigned long long gstep = (unsigned long long)OUT * E * 2; // bytes per s

  auto issue_tdm = [&](unsigned ldsaddr, unsigned long long ga) {
    u32x4 g0;
    g0[0] = 1u;                                   // count=1, user descriptor
    g0[1] = ldsaddr;                              // lds_addr
    g0[2] = (unsigned)ga;                         // global_addr[31:0]
    g0[3] = (unsigned)((ga >> 32) & 0x01FFFFFFu) | 0x80000000u; // addr[56:32] | type=2
    i32x8 g1;
    g1[0] = (int)((1u << 16) | (1u << 20) | (5u << 22) | (3u << 25)); // 2B, pad 4DW/64DW
    g1[1] = (int)(128u << 16);                    // tensor_dim0 = 128 (lo16 @ [63:48])
    g1[2] = (int)(((unsigned)(IN * OUT) & 0xFFFFu) << 16);  // tensor_dim1 lo16
    g1[3] = (int)(((unsigned)(IN * OUT) >> 16) | (128u << 16)); // dim1 hi16 | tile_dim0
    g1[4] = (int)(unsigned)COLS;                  // tile_dim1 = COLS, tile_dim2 = 0
    g1[5] = (int)128;                             // tensor_dim0_stride = E
    g1[6] = (int)(((unsigned)(OUT * E) & 0xFFFFu) << 16);   // tensor_dim1_stride lo16
    g1[7] = (int)((unsigned)(OUT * E) >> 16);               // tensor_dim1_stride hi
    i32x4 g2; g2[0] = 1; g2[1] = 0; g2[2] = 0; g2[3] = 0;   // tensor_dim2=1
    i32x4 g3; g3[0] = 0; g3[1] = 0; g3[2] = 0; g3[3] = 0;
    i32x8 g4; g4[0] = 0; g4[1] = 0; g4[2] = 0; g4[3] = 0;   // unused trailing group
    g4[4] = 0; g4[5] = 0; g4[6] = 0; g4[7] = 0;
    __builtin_amdgcn_tensor_load_to_lds(g0, g1, g2, g3, g4, 0);
  };

  if (wave == 0) issue_tdm(lds0, gaddr);          // prologue: tile for s=0

  for (int s = 0; s < IN; ++s) {
    if (wave == 0) __builtin_amdgcn_s_wait_tensorcnt(0);  // tile s landed
    __syncthreads();                                       // publish; prev buffer free
    if (wave == 0 && s + 1 < IN)
      issue_tdm((s & 1) ? lds0 : lds1, gaddr + gstep);     // prefetch tile s+1
    gaddr += gstep;

    float  xv = xs[s * 16 + row];
    bf16_t xb = (bf16_t)xv;
    v16bf  xsp;
#pragma unroll
    for (int j = 0; j < 16; ++j) xsp[j] = xb;

    const char* base = &bstage[s & 1][(wave * 16 * NT) * ROWB];
#pragma unroll
    for (int c = 0; c < 4; ++c) {
      v16bf afrag = hf[c] * xsp;                  // v_pk_mul_bf16 x8
#pragma unroll
      for (int t = 0; t < NT; ++t) {
        const v16bf bfrag =
            *(const v16bf*)(base + (t * 16 + row) * ROWB + khalf * 32 + c * 64);
        acc[t] = __builtin_amdgcn_wmma_f32_16x16x32_bf16(
            false, afrag, false, bfrag, (short)0, acc[t], false, false);
      }
    }
  }

#pragma unroll
  for (int t = 0; t < NT; ++t)
#pragma unroll
    for (int v = 0; v < 8; ++v) {
      float val = acc[t][v];
      if (RELU) val = val > 0.f ? val : 0.f;
      y[(size_t)(b0 + v + khalf * 8) * OUT + (n0 + t * 16 + row)] = val;
    }
}

// ---------------------------------------------------------------------------
extern "C" void kernel_launch(void* const* d_in, const int* in_sizes, int n_in,
                              void* d_out, int out_size, void* d_ws, size_t ws_size,
                              hipStream_t stream) {
  const float* input = (const float*)d_in[0];
  const float* l1W = (const float*)d_in[1];  const float* l1b = (const float*)d_in[2];
  const float* w1W = (const float*)d_in[3];  const float* w1b = (const float*)d_in[4];
  const float* b1W = (const float*)d_in[5];  const float* b1b = (const float*)d_in[6];
  const float* w2W = (const float*)d_in[7];  const float* w2b = (const float*)d_in[8];
  const float* b2W = (const float*)d_in[9];  const float* b2b = (const float*)d_in[10];
  const float* w3W = (const float*)d_in[11]; const float* w3b = (const float*)d_in[12];
  const float* b3W = (const float*)d_in[13]; const float* b3b = (const float*)d_in[14];
  const float* w4W = (const float*)d_in[15]; const float* w4b = (const float*)d_in[16];
  const float* b4W = (const float*)d_in[17]; const float* b4b = (const float*)d_in[18];
  const float* w5W = (const float*)d_in[19]; const float* w5b = (const float*)d_in[20];
  const float* b5W = (const float*)d_in[21]; const float* b5b = (const float*)d_in[22];

  char* p = (char*)d_ws;
  float*  h    = (float*)p;  p += (size_t)BN * HID * 4;
  float*  x0   = (float*)p;  p += (size_t)BN * STATE * 4;
  float*  bufA = (float*)p;  p += (size_t)BN * 256 * 4;
  float*  bufB = (float*)p;  p += (size_t)BN * 256 * 4;
  float*  BM   = (float*)p;  p += (size_t)BN * 256 * 4;
  bf16_t* wbf  = (bf16_t*)p; p += (size_t)256 * 256 * E * 2 + 65536;  // +64KB pad
  float*  out  = (float*)d_out;

  prep_h<<<(BN * HID + 255) / 256, 256, 0, stream>>>(input, l1W, l1b, h);
  prep_x0<<<(BN * STATE + 255) / 256, 256, 0, stream>>>(input, x0);

  const int W1N4 = STATE * 256 * E / 4, W2N4 = 256 * 256 * E / 4, W5N4 = 256 * 16 * E / 4;

  // layer 1: IN=176 -> 256, chunks iw=0, ib=1, relu
  cvt_bf16<<<(W1N4 + 255) / 256, 256, 0, stream>>>(w1W, wbf, W1N4);
  bias_mat<STATE, 256><<<(BN * 256 + 255) / 256, 256, 0, stream>>>(x0, h + 1 * E, w1b, b1W, b1b, BM);
  layer_wmma<STATE, 256, 1, 2, 2><<<dim3(BN / 16, 4), 64, 0, stream>>>(x0, h + 0 * E, wbf, BM, bufA);

  // layer 2: 256 -> 256, iw=2, ib=3, relu
  cvt_bf16<<<(W2N4 + 255) / 256, 256, 0, stream>>>(w2W, wbf, W2N4);
  bias_mat<256, 256><<<(BN * 256 + 255) / 256, 256, 0, stream>>>(bufA, h + 3 * E, w2b, b2W, b2b, BM);
  layer_wmma<256, 256, 1, 2, 2><<<dim3(BN / 16, 4), 64, 0, stream>>>(bufA, h + 2 * E, wbf, BM, bufB);

  // layer 3: 256 -> 256, iw=4, ib=5, relu
  cvt_bf16<<<(W2N4 + 255) / 256, 256, 0, stream>>>(w3W, wbf, W2N4);
  bias_mat<256, 256><<<(BN * 256 + 255) / 256, 256, 0, stream>>>(bufB, h + 5 * E, w3b, b3W, b3b, BM);
  layer_wmma<256, 256, 1, 2, 2><<<dim3(BN / 16, 4), 64, 0, stream>>>(bufB, h + 4 * E, wbf, BM, bufA);

  // layer 4: 256 -> 256, iw=6, ib=7, NO relu
  cvt_bf16<<<(W2N4 + 255) / 256, 256, 0, stream>>>(w4W, wbf, W2N4);
  bias_mat<256, 256><<<(BN * 256 + 255) / 256, 256, 0, stream>>>(bufA, h + 7 * E, w4b, b4W, b4b, BM);
  layer_wmma<256, 256, 0, 2, 2><<<dim3(BN / 16, 4), 64, 0, stream>>>(bufA, h + 6 * E, wbf, BM, bufB);

  // layer 5: 256 -> 16, iw=8, ib=9, relu  (writes d_out directly)
  cvt_bf16<<<(W5N4 + 255) / 256, 256, 0, stream>>>(w5W, wbf, W5N4);
  bias_mat<256, 16><<<(BN * 16 + 255) / 256, 256, 0, stream>>>(bufB, h + 9 * E, w5b, b5W, b5b, BM);
  layer_wmma<256, 16, 1, 1, 1><<<dim3(BN / 16, 1), 32, 0, stream>>>(bufB, h + 8 * E, wbf, BM, out);
}
